// CausalWanSelfAttention_22479858827927
// MI455X (gfx1250) — compile-verified, matrix operand order checked by reading, not run on hardware
//
#include <hip/hip_runtime.h>
#include <hip/hip_bf16.h>

typedef __attribute__((ext_vector_type(16))) __bf16 v16bf;
typedef __attribute__((ext_vector_type(8)))  float  v8f;
typedef __attribute__((ext_vector_type(2)))  __bf16 bf16x2;
typedef __attribute__((ext_vector_type(4)))  unsigned int u32x4;
typedef __attribute__((ext_vector_type(8)))  int i32x8;
typedef __attribute__((ext_vector_type(4)))  int i32x4;
typedef __bf16 bf16;

// Problem constants (fixed by the reference)
constexpr int Ltok = 2688;   // 6*16*28
constexpr int Dm   = 1536;
constexpr int NHn  = 12;
constexpr int HDd  = 128;
constexpr int Hh   = 16;
constexpr int Ww   = 28;
constexpr int Blk  = 1344;   // 3*16*28

// GEMM tiling
constexpr int TM = 128, TN = 128, TK = 64;

// ---------------------------------------------------------------------------
// Fragment helpers.  CDNA5 16-bit A-matrix 16x32 layout (ISA 7.12.2).
// ---------------------------------------------------------------------------
__device__ __forceinline__ v16bf load_frag_row(const bf16* p, int hi) {
  v16bf r;
#pragma unroll
  for (int v = 0; v < 8; ++v) {
    int kb = ((v < 4) ? (2 * v) : (16 + 2 * (v - 4))) + hi * 8;
    bf16x2 t = *reinterpret_cast<const bf16x2*>(p + kb);
    r[2 * v]     = t[0];
    r[2 * v + 1] = t[1];
  }
  return r;
}

__device__ __forceinline__ v8f wmma_bf16(v16bf a, v16bf b, v8f c) {
  return __builtin_amdgcn_wmma_f32_16x16x32_bf16(false, a, false, b, (short)0, c,
                                                 false, false);
}

// CDNA5 async DMA: LDS[lds_off] = MEM[saddr + voff] (16 bytes/lane, ASYNCcnt).
__device__ __forceinline__ void async_b128(unsigned lds_off, unsigned voff,
                                           const void* base, const void* lds_base) {
  asm volatile("global_load_async_to_lds_b128 %0, %1, %2 offset:0"
               :: "v"(lds_off), "v"(voff), "s"(base), "v"(lds_base)
               : "memory");
}
__device__ __forceinline__ void wait_async0() {
  asm volatile("s_wait_asynccnt 0" ::: "memory");
}
// Compiler-level LDS fence + escape: forbids folding away ds reads of smem
// written behind the compiler's back (TDM / async DMA).
__device__ __forceinline__ void fence_lds(const void* p) {
  asm volatile("" :: "v"(p) : "memory");
}

// ---------------------------------------------------------------------------
// Tensor Data Mover: 2-D tile load Global -> LDS (TENSORcnt-tracked).
// D# packing per ISA 8.3/8.4; data_size = 2 bytes; groups 2/3 zero (2-D).
// amdgpu-toolchain builtin: (u32x4, i32x8, i32x4, i32x4, i32x8, i32 cpol).
// ---------------------------------------------------------------------------
__device__ __forceinline__ void tdm_load_2d(unsigned lds_off, const void* gaddr,
                                            unsigned tile_x, unsigned tile_y,
                                            unsigned tensor_x, unsigned tensor_y,
                                            unsigned stride_x /*elems*/) {
  unsigned long long ga = (unsigned long long)gaddr;
  u32x4 g0;
  g0[0] = 1u;                                             // count=1 (valid D#)
  g0[1] = lds_off;                                        // lds_addr (bytes)
  g0[2] = (unsigned)(ga & 0xffffffffu);                   // global_addr[31:0]
  g0[3] = (unsigned)((ga >> 32) & 0x01ffffffu) | (2u << 30);  // [56:32] | type=2
  i32x8 g1;
  g1[0] = (int)(1u << 16);                                // data_size=2B
  g1[1] = (int)((tensor_x & 0xffffu) << 16);              // dim0[15:0]
  g1[2] = (int)(((tensor_x >> 16) & 0xffffu) | ((tensor_y & 0xffffu) << 16));
  g1[3] = (int)(((tensor_y >> 16) & 0xffffu) | ((tile_x & 0xffffu) << 16));
  g1[4] = (int)(tile_y & 0xffffu);                        // tile_dim1; dim2=0
  g1[5] = (int)stride_x;                                  // dim0_stride[31:0]
  g1[6] = 0;                                              // stride hi, dim1_stride
  g1[7] = 0;
  i32x4 z4 = {0, 0, 0, 0};
  i32x8 z8 = {0, 0, 0, 0, 0, 0, 0, 0};
  __builtin_amdgcn_tensor_load_to_lds(g0, g1, z4, z4, z8, 0);
}
__device__ __forceinline__ void wait_tensor0() {
  __builtin_amdgcn_s_wait_tensorcnt((short)0);
}

// ---------------------------------------------------------------------------
// f32 -> bf16 convert
// ---------------------------------------------------------------------------
__global__ void cvt_f32_bf16(const float* __restrict__ in, bf16* __restrict__ out,
                             int n) {
  int i = blockIdx.x * blockDim.x + threadIdx.x;
  if (i < n) out[i] = (bf16)in[i];
}

// Transpose + convert a DIMxDIM weight: WT[n][k] = W[k][n]
__global__ void transpose_cvt(const float* __restrict__ W, bf16* __restrict__ WT) {
  int n = blockIdx.x * 16 + threadIdx.x;
  int k = blockIdx.y * 16 + threadIdx.y;
  WT[(size_t)n * Dm + k] = (bf16)W[(size_t)k * Dm + n];
}

// ---------------------------------------------------------------------------
// C[M,N] = A[M,K]@Bt[N,K]^T + bias  (A,Bt bf16, C f32)
// 256 threads (8 waves), 128x128 tile, K-step 64.
// Double-buffered LDS staged by the Tensor Data Mover: wave 0 issues one
// descriptor per A/B panel; everyone else just computes.
// ---------------------------------------------------------------------------
__global__ __launch_bounds__(256) void gemm_bf16_nt(
    const bf16* __restrict__ A, const bf16* __restrict__ Bt,
    const float* __restrict__ bias, float* __restrict__ C,
    int M, int N, int K) {
  __shared__ __align__(16) bf16 sm[2][2][TM * TK];  // [buf][A|B] = 64 KB
  const void* smBase = (const void*)&sm[0][0][0];

  const int tid  = threadIdx.x;
  const int wave = tid >> 5;
  const int lane = tid & 31;
  const int hi   = lane >> 4;
  const int col  = lane & 15;
  const int wm   = wave & 3;
  const int wn   = wave >> 2;
  const int m0   = blockIdx.x * TM;
  const int n0   = blockIdx.y * TN;

  v8f acc[2][4];
#pragma unroll
  for (int mi = 0; mi < 2; ++mi)
#pragma unroll
    for (int ni = 0; ni < 4; ++ni) acc[mi][ni] = {};

  // TDM: stage the 128x64 A panel and 128x64 B panel of this K-step.
  auto stage = [&](int k0, int buf) {
    unsigned loA = (unsigned)((buf * 2 + 0) * (TM * TK) * 2);
    tdm_load_2d(loA, A + (size_t)m0 * K + k0,
                (unsigned)TK, (unsigned)TM, (unsigned)K, (unsigned)M,
                (unsigned)K);
    unsigned loB = (unsigned)((buf * 2 + 1) * (TM * TK) * 2);
    tdm_load_2d(loB, Bt + (size_t)n0 * K + k0,
                (unsigned)TK, (unsigned)TN, (unsigned)K, (unsigned)N,
                (unsigned)K);
  };

  if (wave == 0) {
    stage(0, 0);
    wait_tensor0();
  }
  fence_lds(smBase);
  __syncthreads();

  int buf = 0;
  const int NK = K / TK;
  for (int it = 0; it < NK; ++it) {
    if (it + 1 < NK && wave == 0) stage((it + 1) * TK, buf ^ 1);  // overlap

    const bf16* sA = &sm[buf][0][0];
    const bf16* sB = &sm[buf][1][0];
#pragma unroll
    for (int ks = 0; ks < 2; ++ks) {
      v16bf af0 = load_frag_row(sA + (wm * 32 + 0  + col) * TK + ks * 32, hi);
      v16bf af1 = load_frag_row(sA + (wm * 32 + 16 + col) * TK + ks * 32, hi);
#pragma unroll
      for (int ni = 0; ni < 4; ++ni) {
        v16bf bfb = load_frag_row(sB + (wn * 64 + ni * 16 + col) * TK + ks * 32, hi);
        acc[0][ni] = wmma_bf16(af0, bfb, acc[0][ni]);
        acc[1][ni] = wmma_bf16(af1, bfb, acc[1][ni]);
      }
    }
    if (wave == 0) wait_tensor0();  // next tile fully in LDS
    fence_lds(smBase);
    __syncthreads();
    buf ^= 1;
  }

#pragma unroll
  for (int mi = 0; mi < 2; ++mi)
#pragma unroll
    for (int ni = 0; ni < 4; ++ni) {
      int n = n0 + wn * 64 + ni * 16 + col;
      float bval = bias[n];
#pragma unroll
      for (int r = 0; r < 8; ++r) {
        int m = m0 + wm * 32 + mi * 16 + r + hi * 8;
        C[(size_t)m * N + n] = acc[mi][ni][r] + bval;
      }
    }
}

// ---------------------------------------------------------------------------
// Fused RMSNorm (over DIM) + RoPE + bf16 pack.
//   qb,kb : [NH][L][HD]   vt : [NH][HD][L]  (transposed for PV B-fragments)
// ---------------------------------------------------------------------------
__global__ void norm_rope_pack(const float* __restrict__ qf, const float* __restrict__ kf,
                               const float* __restrict__ vf, const float* __restrict__ gq,
                               const float* __restrict__ gk, const float* __restrict__ freqs,
                               bf16* __restrict__ qb, bf16* __restrict__ kb,
                               bf16* __restrict__ vt) {
  const int t   = blockIdx.x;
  const int tid = threadIdx.x;
  __shared__ float sA[256], sB[256];

  const float* qrow = qf + (size_t)t * Dm;
  const float* krow = kf + (size_t)t * Dm;
  float sq = 0.f, sk = 0.f;
  for (int i = tid; i < Dm; i += 256) {
    float a = qrow[i]; sq += a * a;
    float b = krow[i]; sk += b * b;
  }
  sA[tid] = sq; sB[tid] = sk;
  __syncthreads();
  for (int s = 128; s > 0; s >>= 1) {
    if (tid < s) { sA[tid] += sA[tid + s]; sB[tid] += sB[tid + s]; }
    __syncthreads();
  }
  const float rq = rsqrtf(sA[0] / (float)Dm + 1e-6f);
  const float rk = rsqrtf(sB[0] / (float)Dm + 1e-6f);

  const int f   = t / (Hh * Ww);
  const int rem = t % (Hh * Ww);
  const int hh  = rem / Ww;
  const int ww  = rem % Ww;

  for (int pp = tid; pp < Dm / 2; pp += 256) {
    int head = pp >> 6;
    int c    = pp & 63;
    int dim  = head * HDd + 2 * c;
    int pos  = (c < 22) ? f : ((c < 43) ? hh : ww);
    float cv = freqs[((size_t)pos * 64 + c) * 2 + 0];
    float sv = freqs[((size_t)pos * 64 + c) * 2 + 1];
    float qre = qrow[dim]     * rq * gq[dim];
    float qim = qrow[dim + 1] * rq * gq[dim + 1];
    float kre = krow[dim]     * rk * gk[dim];
    float kim = krow[dim + 1] * rk * gk[dim + 1];
    size_t o = ((size_t)head * Ltok + t) * HDd + 2 * c;
    qb[o]     = (bf16)(qre * cv - qim * sv);
    qb[o + 1] = (bf16)(qre * sv + qim * cv);
    kb[o]     = (bf16)(kre * cv - kim * sv);
    kb[o + 1] = (bf16)(kre * sv + kim * cv);
  }
  const float* vrow = vf + (size_t)t * Dm;
  for (int i = tid; i < Dm; i += 256) {
    int head = i >> 7;
    int d    = i & 127;
    vt[((size_t)head * HDd + d) * Ltok + t] = (bf16)vrow[i];
  }
}

// ---------------------------------------------------------------------------
// Flash attention: 128 threads (4 waves); each wave owns one 16-query tile;
// the 4 tiles share one causal block (1344 = 21*64), so the workgroup
// cooperatively streams K/V tiles into double-buffered LDS via async DMA.
// S^T = K.Q^T (lane owns a query column -> lane-local softmax); P^T maps
// directly into the PV WMMA's A-fragment layout.
// ---------------------------------------------------------------------------
__global__ __launch_bounds__(128) void attn_kernel(
    const bf16* __restrict__ qb, const bf16* __restrict__ kb,
    const bf16* __restrict__ vt, bf16* __restrict__ ao) {
  __shared__ __align__(16) bf16 smem[2][32 * HDd + HDd * 32];  // 32 KB
  const void* smBase = (const void*)&smem[0][0];

  const int tid  = threadIdx.x;
  const int wave = tid >> 5;
  const int lane = tid & 31;
  const int hi   = lane >> 4;
  const int qcol = lane & 15;
  const int h    = blockIdx.y;
  const int q0   = (blockIdx.x * 4 + wave) * 16;
  const int kend = ((blockIdx.x * 64) / Blk + 1) * Blk;  // uniform in workgroup

  const bf16* kb_h = kb + (size_t)h * Ltok * HDd;
  const bf16* vt_h = vt + (size_t)h * HDd * Ltok;

  auto stageKV = [&](int j0, int buf) {
    unsigned bufOff = (unsigned)buf * (unsigned)((32 * HDd + HDd * 32) * 2);
#pragma unroll
    for (int i = 0; i < 4; ++i) {
      int c = tid + i * 128;              // 0..511
      int row = c >> 4, seg = c & 15;
      unsigned lo = bufOff + (unsigned)(row * 256 + seg * 16);
      unsigned go = (unsigned)((j0 + row) * HDd * 2 + seg * 16);
      async_b128(lo, go, kb_h, smBase);
    }
#pragma unroll
    for (int i = 0; i < 4; ++i) {
      int c = tid + i * 128;              // 0..511
      int row = c >> 2, seg = c & 3;
      unsigned lo = bufOff + (unsigned)(32 * HDd * 2) + (unsigned)(row * 64 + seg * 16);
      unsigned go = (unsigned)((row * Ltok + j0) * 2 + seg * 16);
      async_b128(lo, go, vt_h, smBase);
    }
  };

  const bf16* qbase = qb + ((size_t)h * Ltok + q0 + qcol) * HDd;
  v16bf qfrag[4];
#pragma unroll
  for (int kk = 0; kk < 4; ++kk) qfrag[kk] = load_frag_row(qbase + kk * 32, hi);

  v8f acc[8];
#pragma unroll
  for (int i = 0; i < 8; ++i) acc[i] = {};
  float mrun = -1e30f, lrun = 0.f;
  const float sc = 0.08838834764831845f;  // 1/sqrt(128)

  stageKV(0, 0);
  wait_async0();
  __syncthreads();

  int buf = 0;
  const int nj = kend / 32;
  for (int jt = 0; jt < nj; ++jt) {
    if (jt + 1 < nj) stageKV((jt + 1) * 32, buf ^ 1);

    const bf16* sKp = &smem[buf][0];
    const bf16* sVp = &smem[buf][32 * HDd];

    v8f st0 = {}, st1 = {};
#pragma unroll
    for (int kk = 0; kk < 4; ++kk)
      st0 = wmma_bf16(load_frag_row(sKp + (0  + qcol) * HDd + kk * 32, hi),
                      qfrag[kk], st0);
#pragma unroll
    for (int kk = 0; kk < 4; ++kk)
      st1 = wmma_bf16(load_frag_row(sKp + (16 + qcol) * HDd + kk * 32, hi),
                      qfrag[kk], st1);

    float s0[8], s1[8];
    float mloc = -1e30f;
#pragma unroll
    for (int r = 0; r < 8; ++r) {
      s0[r] = st0[r] * sc; s1[r] = st1[r] * sc;
      mloc = fmaxf(mloc, fmaxf(s0[r], s1[r]));
    }
    mloc = fmaxf(mloc, __shfl_xor(mloc, 16));
    float mnew = fmaxf(mrun, mloc);
    float sfac = __expf(mrun - mnew);
    float tsum = 0.f;
    v16bf ap;  // P^T A-fragment: elems 0..7 <- st0 rows, 8..15 <- st1 rows
#pragma unroll
    for (int r = 0; r < 8; ++r) {
      float p0 = __expf(s0[r] - mnew);
      float p1 = __expf(s1[r] - mnew);
      tsum += p0 + p1;
      ap[r]     = (bf16)p0;
      ap[8 + r] = (bf16)p1;
    }
    tsum += __shfl_xor(tsum, 16);
    lrun = lrun * sfac + tsum;
    mrun = mnew;

    float sr[8];
#pragma unroll
    for (int r = 0; r < 8; ++r) sr[r] = __shfl(sfac, r + hi * 8);

#pragma unroll
    for (int nt = 0; nt < 8; ++nt) {
#pragma unroll
      for (int r = 0; r < 8; ++r) acc[nt][r] *= sr[r];
      acc[nt] = wmma_bf16(ap, load_frag_row(sVp + (nt * 16 + qcol) * 32, hi),
                          acc[nt]);
    }

    wait_async0();
    __syncthreads();
    buf ^= 1;
  }

  float inv = 1.f / lrun;
  float ir[8];
#pragma unroll
  for (int r = 0; r < 8; ++r) ir[r] = __shfl(inv, r + hi * 8);
#pragma unroll
  for (int nt = 0; nt < 8; ++nt) {
    int d = h * HDd + nt * 16 + qcol;
#pragma unroll
    for (int r = 0; r < 8; ++r) {
      int q = q0 + r + hi * 8;
      ao[(size_t)q * Dm + d] = (bf16)(acc[nt][r] * ir[r]);
    }
  }
}

// ---------------------------------------------------------------------------
extern "C" void kernel_launch(void* const* d_in, const int* in_sizes, int n_in,
                              void* d_out, int out_size, void* d_ws, size_t ws_size,
                              hipStream_t stream) {
  (void)in_sizes; (void)n_in; (void)out_size; (void)ws_size;
  const float* x     = (const float*)d_in[0];
  const float* freqs = (const float*)d_in[3];
  const float* Wq = (const float*)d_in[4];
  const float* bq = (const float*)d_in[5];
  const float* Wk = (const float*)d_in[6];
  const float* bk = (const float*)d_in[7];
  const float* Wv = (const float*)d_in[8];
  const float* bv = (const float*)d_in[9];
  const float* Wo = (const float*)d_in[10];
  const float* bo = (const float*)d_in[11];
  const float* gq = (const float*)d_in[12];
  const float* gk = (const float*)d_in[13];
  float* out = (float*)d_out;

  char* ws = (char*)d_ws;
  size_t off = 0;
  auto carve = [&](size_t bytes) {
    void* p = ws + off;
    off += (bytes + 255) & ~(size_t)255;
    return p;
  };
  constexpr size_t XD  = (size_t)Ltok * Dm;
  constexpr size_t WW2 = (size_t)Dm * Dm;
  bf16* xb  = (bf16*)carve(XD * 2);
  bf16* WqT = (bf16*)carve(WW2 * 2);
  bf16* WkT = (bf16*)carve(WW2 * 2);
  bf16* WvT = (bf16*)carve(WW2 * 2);
  bf16* WoT = (bf16*)carve(WW2 * 2);
  float* qfb = (float*)carve(XD * 4);
  float* kfb = (float*)carve(XD * 4);
  float* vfb = (float*)carve(XD * 4);
  bf16* qbb = (bf16*)carve(XD * 2);
  bf16* kbb = (bf16*)carve(XD * 2);
  bf16* vtt = (bf16*)carve(XD * 2);
  bf16* aob = (bf16*)carve(XD * 2);

  {
    int n = (int)XD;
    cvt_f32_bf16<<<(n + 255) / 256, 256, 0, stream>>>(x, xb, n);
  }
  dim3 tg(Dm / 16, Dm / 16), tb(16, 16);
  transpose_cvt<<<tg, tb, 0, stream>>>(Wq, WqT);
  transpose_cvt<<<tg, tb, 0, stream>>>(Wk, WkT);
  transpose_cvt<<<tg, tb, 0, stream>>>(Wv, WvT);
  transpose_cvt<<<tg, tb, 0, stream>>>(Wo, WoT);

  dim3 gg(Ltok / TM, Dm / TN);  // (21, 12)
  gemm_bf16_nt<<<gg, 256, 0, stream>>>(xb, WqT, bq, qfb, Ltok, Dm, Dm);
  gemm_bf16_nt<<<gg, 256, 0, stream>>>(xb, WkT, bk, kfb, Ltok, Dm, Dm);
  gemm_bf16_nt<<<gg, 256, 0, stream>>>(xb, WvT, bv, vfb, Ltok, Dm, Dm);

  norm_rope_pack<<<Ltok, 256, 0, stream>>>(qfb, kfb, vfb, gq, gk, freqs,
                                           qbb, kbb, vtt);

  attn_kernel<<<dim3(Ltok / 64, NHn), 128, 0, stream>>>(qbb, kbb, vtt, aob);

  gemm_bf16_nt<<<gg, 256, 0, stream>>>(aob, WoT, bo, out, Ltok, Dm, Dm);
}
